// VisualAttentionLayer_43885975830533
// MI455X (gfx1250) — matrix operand order repaired and checked by reading, not compiled, not run on hardware
//
#include <hip/hip_runtime.h>
#include <hip/hip_fp16.h>

typedef __attribute__((ext_vector_type(16))) _Float16 v16h;
typedef __attribute__((ext_vector_type(8)))  float    v8f;

#define BATCH   32
#define IMG     512
#define OUT_HW  64
#define CHN     3
#define N1      (IMG * CHN)   // 1536, folded (w,c) axis
#define G1_NT   128
#define KT      32

#define G1_AST  68            // A tile [k2][m], 64 + 4 pad dwords
#define G1_BST  132           // B tile [k2][n], 128 + 4 pad dwords
#define G2_ST   68            // GEMM2 tiles [k2][64], 64 + 4 pad

union FragH { v16h v; uint32_t u[8]; };

// ---- WMMA fragment loaders from K-major LDS tiles (wave32, 16x16x32 f16) ----
// A-matrix 16x32: lane L holds M = m0 + (L&15); VGPR j -> k2 = j + 8*(L>>4).
// Tile layout: [k2][m], row stride ST.
template<int ST>
__device__ __forceinline__ v16h load_a_km(const uint32_t* Ak, int m0, int lane) {
    const int hw = lane >> 4;
    const int m  = m0 + (lane & 15);
    FragH f;
    const uint32_t* p = Ak + (8 * hw) * ST + m;
#pragma unroll
    for (int j = 0; j < 8; ++j) f.u[j] = p[j * ST];
    return f.v;
}
// B-matrix 32x16: lane L holds N = n0 + (L&15); VGPR j -> k2 = j + 8*(L>>4).
template<int ST>
__device__ __forceinline__ v16h load_b_km(const uint32_t* Bk, int n0, int lane) {
    const int hw = lane >> 4;
    const int n  = n0 + (lane & 15);
    FragH f;
    const uint32_t* p = Bk + (8 * hw) * ST + n;
#pragma unroll
    for (int j = 0; j < 8; ++j) f.u[j] = p[j * ST];
    return f.v;
}

__device__ __forceinline__ uint32_t pack2f(float a, float b) {
    union { __half2 h; uint32_t u; } cv;
    cv.h = __halves2half2(__float2half(a), __float2half(b));
    return cv.u;
}
__device__ __forceinline__ uint32_t pack2h(__half a, __half b) {
    union { __half2 h; uint32_t u; } cv;
    cv.h = __halves2half2(a, b);
    return cv.u;
}

// ---------------------------------------------------------------------------
// Kernel 1: Gaussian filterbanks. One wave32 per (b, axis, r) column of 512.
// ---------------------------------------------------------------------------
__global__ __launch_bounds__(256)
void va_mask_kernel(const float* __restrict__ params,
                    __half* __restrict__ Ayt,   // [B][64][512]
                    __half* __restrict__ Axh)   // [B][512][64]
{
    const int wid  = blockIdx.x * (blockDim.x >> 5) + (threadIdx.x >> 5);
    const int lane = threadIdx.x & 31;
    const int b    = wid >> 7;
    const int rem  = wid & 127;
    const int axis = rem >> 6;
    const int r    = rem & 63;

    const float* p = params + b * 6 + axis * 3;   // (u, s, d)
    const float centre = p[0] + (float)r * p[2];
    const float inv_s  = 1.0f / p[1];

    float vals[16];
    float sum = 0.0f;
#pragma unroll
    for (int i = 0; i < 16; ++i) {
        const float c = (float)(lane + i * 32);
        const float z = (c - centre) * inv_s;
        const float e = __expf(-0.5f * z * z);
        vals[i] = e;
        sum += e;
    }
#pragma unroll
    for (int m = 16; m >= 1; m >>= 1) sum += __shfl_xor(sum, m, 32);
    const float inv = 1.0f / (sum + 1e-8f);

    if (axis == 0) {
        __half* row = Ayt + ((size_t)b * 64 + r) * 512;
#pragma unroll
        for (int i = 0; i < 16; ++i)
            row[lane + i * 32] = __float2half(vals[i] * inv);
    } else {
        __half* base = Axh + (size_t)b * 512 * 64 + r;
#pragma unroll
        for (int i = 0; i < 16; ++i)
            base[(size_t)(lane + i * 32) * 64] = __float2half(vals[i] * inv);
    }
}

// ---------------------------------------------------------------------------
// Kernel 2: g[b][H][x] = sum_h Ayt[b][H][h] * img[b][h][x],  x = w*3+c
// M=64, K=512, N=1536. 256 thr (8 waves), tile 64x128, KT=32, double-buffered.
// ---------------------------------------------------------------------------
__global__ __launch_bounds__(256)
void va_gemm1_kernel(const float* __restrict__ img,   // [B][512][1536]
                     const __half* __restrict__ Ayt,  // [B][64][512]
                     __half* __restrict__ g)          // [B][64][1536]
{
    __shared__ uint32_t Asl[2][16 * G1_AST];   // 2 x 4.25 KB
    __shared__ uint32_t Bsl[2][16 * G1_BST];   // 2 x 8.25 KB

    const int b    = blockIdx.y;
    const int x0   = blockIdx.x * G1_NT;
    const int t    = threadIdx.x;
    const int lane = t & 31;
    const int wave = t >> 5;

    const float*  imgB = img + (size_t)b * IMG * N1;
    const __half* A    = Ayt + (size_t)b * 64 * 512;

    // Stage mask tile 64x32 f16: uint4 row read = 4 ready-packed K-pairs.
    auto stageA = [&](uint32_t* As, int k0) {
        const int m = t >> 2, seg = t & 3;
        const uint4 av = *(const uint4*)(A + (size_t)m * 512 + k0 + seg * 8);
        As[(seg * 4 + 0) * G1_AST + m] = av.x;
        As[(seg * 4 + 1) * G1_AST + m] = av.y;
        As[(seg * 4 + 2) * G1_AST + m] = av.z;
        As[(seg * 4 + 3) * G1_AST + m] = av.w;
    };
    // Stage image tile 32x128 f32 -> f16 K-pairs; one b128 LDS store per unit.
    auto stageB = [&](uint32_t* Bs, int k0) {
#pragma unroll
        for (int it = 0; it < 2; ++it) {
            const int u  = t + it * 256;          // 512 units: [k2][n4]
            const int n  = (u & 31) * 4;
            const int k2 = u >> 5;
            const float* p0 = imgB + (size_t)(k0 + 2 * k2) * N1 + x0 + n;
            const float4 f0 = *(const float4*)p0;
            const float4 f1 = *(const float4*)(p0 + N1);
            uint32_t* d = Bs + k2 * G1_BST + n;   // 16B-aligned, contiguous
            d[0] = pack2f(f0.x, f1.x);
            d[1] = pack2f(f0.y, f1.y);
            d[2] = pack2f(f0.z, f1.z);
            d[3] = pack2f(f0.w, f1.w);
        }
    };

    v8f acc[4] = {v8f{}, v8f{}, v8f{}, v8f{}};

    stageA(Asl[0], 0);
    stageB(Bsl[0], 0);
    __syncthreads();

    int buf = 0;
    for (int k0 = 0; k0 < IMG; k0 += KT, buf ^= 1) {
        if (k0 + KT < IMG) {                 // prefetch next tile into other buf
            stageA(Asl[buf ^ 1], k0 + KT);
            stageB(Bsl[buf ^ 1], k0 + KT);
        }
        const v16h bf = load_b_km<G1_BST>(Bsl[buf], wave * 16, lane);
#pragma unroll
        for (int mt = 0; mt < 4; ++mt) {
            const v16h af = load_a_km<G1_AST>(Asl[buf], mt * 16, lane);
            acc[mt] = __builtin_amdgcn_wmma_f32_16x16x32_f16(
                false, af, false, bf, (short)0, acc[mt], false, false);
        }
        __syncthreads();
    }

    const int hw   = lane >> 4;
    const int ncol = x0 + wave * 16 + (lane & 15);
    __half* gB = g + (size_t)b * 64 * N1 + ncol;
#pragma unroll
    for (int mt = 0; mt < 4; ++mt)
#pragma unroll
        for (int v = 0; v < 8; ++v) {
            const int m = mt * 16 + v + 8 * hw;
            gB[(size_t)m * N1] = __float2half(acc[mt][v]);
        }
}

// ---------------------------------------------------------------------------
// Kernel 3: out[b][H][W][c] = sum_w g[b][H][w*3+c] * Axh[b][w][W]
// One block per (c, b). M=64, N=64, K=512, double-buffered.
// ---------------------------------------------------------------------------
__global__ __launch_bounds__(256)
void va_gemm2_kernel(const __half* __restrict__ g,    // [B][64][1536]
                     const __half* __restrict__ Axh,  // [B][512][64]
                     float* __restrict__ out)         // [B][64][64][3]
{
    __shared__ uint32_t Asl[2][16 * G2_ST];
    __shared__ uint32_t Bsl[2][16 * G2_ST];

    const int c = blockIdx.x, b = blockIdx.y;
    const int t = threadIdx.x, lane = t & 31, wave = t >> 5;
    const int m0w = (wave & 3) * 16;
    const int n0w = (wave >> 2) * 32;

    const __half* gB = g   + (size_t)b * 64 * N1;
    const __half* Ax = Axh + (size_t)b * 512 * 64;

    // A tile: stride-3 channel pick from g, packed along K into [k2][m].
    auto stageA = [&](uint32_t* As, int k0) {
#pragma unroll
        for (int it = 0; it < 4; ++it) {
            const int u  = t + it * 256;          // 1024 units: [m][k2]
            const int m  = u >> 4;
            const int k2 = u & 15;
            const int k  = k0 + 2 * k2;
            As[k2 * G2_ST + m] =
                pack2h(gB[(size_t)m * N1 + k * CHN + c],
                       gB[(size_t)m * N1 + (k + 1) * CHN + c]);
        }
    };
    // B tile: Ax rows packed along K into [k2][n]; coalesced over n.
    auto stageB = [&](uint32_t* Bs, int k0) {
#pragma unroll
        for (int it = 0; it < 4; ++it) {
            const int u  = t + it * 256;          // 1024 units: [k2][n]
            const int k2 = u >> 6;
            const int n  = u & 63;
            const int k  = k0 + 2 * k2;
            Bs[k2 * G2_ST + n] = pack2h(Ax[(size_t)k * 64 + n],
                                        Ax[(size_t)(k + 1) * 64 + n]);
        }
    };

    v8f acc[2] = {v8f{}, v8f{}};

    stageA(Asl[0], 0);
    stageB(Bsl[0], 0);
    __syncthreads();

    int buf = 0;
    for (int k0 = 0; k0 < IMG; k0 += KT, buf ^= 1) {
        if (k0 + KT < IMG) {
            stageA(Asl[buf ^ 1], k0 + KT);
            stageB(Bsl[buf ^ 1], k0 + KT);
        }
        const v16h af = load_a_km<G2_ST>(Asl[buf], m0w, lane);
#pragma unroll
        for (int nt = 0; nt < 2; ++nt) {
            const v16h bf = load_b_km<G2_ST>(Bsl[buf], n0w + nt * 16, lane);
            acc[nt] = __builtin_amdgcn_wmma_f32_16x16x32_f16(
                false, af, false, bf, (short)0, acc[nt], false, false);
        }
        __syncthreads();
    }

    const int hw = lane >> 4;
    float* ob = out + (size_t)b * OUT_HW * OUT_HW * CHN + c;
#pragma unroll
    for (int nt = 0; nt < 2; ++nt)
#pragma unroll
        for (int v = 0; v < 8; ++v) {
            const int m = m0w + v + 8 * hw;
            const int n = n0w + nt * 16 + (lane & 15);
            ob[(size_t)(m * OUT_HW + n) * CHN] = acc[nt][v];
        }
}

extern "C" void kernel_launch(void* const* d_in, const int* in_sizes, int n_in,
                              void* d_out, int out_size, void* d_ws, size_t ws_size,
                              hipStream_t stream) {
    (void)in_sizes; (void)n_in; (void)out_size; (void)ws_size;
    const float* img    = (const float*)d_in[0];   // (32,512,512,3) f32
    const float* params = (const float*)d_in[1];   // (32,6) f32
    float* out = (float*)d_out;                    // (32,64,64,3) f32

    // workspace: Ayt (2 MB) | Axh (2 MB) | g (6 MB)
    char* ws = (char*)d_ws;
    const size_t maskBytes = (size_t)BATCH * 64 * 512 * sizeof(__half);
    __half* Ayt  = (__half*)ws;
    __half* Axh  = (__half*)(ws + maskBytes);
    __half* gbuf = (__half*)(ws + 2 * maskBytes);

    va_mask_kernel<<<512, 256, 0, stream>>>(params, Ayt, Axh);

    dim3 grid1(N1 / G1_NT, BATCH);   // (12, 32)
    va_gemm1_kernel<<<grid1, 256, 0, stream>>>(img, Ayt, gbuf);

    dim3 grid2(CHN, BATCH);          // (3, 32)
    va_gemm2_kernel<<<grid2, 256, 0, stream>>>(gbuf, Axh, out);
}
